// PartLayer_24524263260457
// MI455X (gfx1250) — matrix operand order repaired and checked by reading, not compiled, not run on hardware
//
#include <hip/hip_runtime.h>

// ---------------- CDNA5 WMMA types / helpers (wave32) ----------------
typedef __attribute__((ext_vector_type(16))) __bf16 v16bf;
typedef __attribute__((ext_vector_type(8)))  float  v8f;

__device__ __forceinline__ v8f wmma_bf16f32(v16bf a, v16bf b, v8f c) {
  // emits v_wmma_f32_16x16x32_bf16
  return __builtin_amdgcn_wmma_f32_16x16x32_bf16(false, a, false, b, (short)0, c,
                                                 false, false);
}

// A (16x32 bf16) fragment from row-major LDS: elems 0..15 at k0+khalf*16 (16B
// contiguous chunks), elems 8..15 at +8 elems. Both chunks 16B aligned by layout.
__device__ __forceinline__ v16bf load_a_lds16(const __bf16* rowbase, int k0, int khalf) {
  const __bf16* p = rowbase + k0 + khalf * 16;
  union { v16bf v; uint4 q[2]; } u;
  u.q[0] = *(const uint4*)(p);        // ds_load_b128
  u.q[1] = *(const uint4*)(p + 8);    // ds_load_b128
  return u.v;
}

// B (32x16 bf16) fragment when the 16 K-values are contiguous in memory
// (column-major J, or a row of k^T). p must be 16B aligned.
__device__ __forceinline__ v16bf load_b_contig(const __bf16* p) {
  union { v16bf v; uint4 q[2]; } u;
  u.q[0] = *(const uint4*)(p);        // ds_load_b128
  u.q[1] = *(const uint4*)(p + 8);    // ds_load_b128
  return u.v;
}

// A fragment from global f32, fully in-bounds: unconditional loads.
__device__ __forceinline__ v16bf load_a_g32_fast(const float* p, int ld, int row,
                                                 int k0, int khalf) {
  const float* base = p + (size_t)row * ld + k0 + khalf * 16;
  v16bf a;
#pragma unroll
  for (int e = 0; e < 8; ++e) a[e] = (__bf16)base[e];
#pragma unroll
  for (int e = 0; e < 8; ++e) a[8 + e] = (__bf16)base[8 + e];
  return a;
}

// A fragment from global f32 with branch-free zero padding: clamp the index so
// the load is always legal, then v_cndmask the result (no exec-mask branches).
__device__ __forceinline__ v16bf load_a_g32_guard(const float* p, int ld, int rows,
                                                  int cols, int row, int k0, int khalf) {
  const bool rok = row < rows;
  const float* base = p + (size_t)(rok ? row : 0) * ld;
  v16bf a;
#pragma unroll
  for (int e = 0; e < 16; ++e) {
    int k = k0 + khalf * 16 + e;
    bool ok = rok && (k < cols);
    float v = base[ok ? k : 0];
    a[e] = (__bf16)(ok ? v : 0.f);
  }
  return a;
}

// B fragment from global f32 weight matrix [K x ld], all indices in bounds
__device__ __forceinline__ v16bf load_b_g32(const float* p, int ld, int k0,
                                            int col, int khalf) {
  v16bf b;
#pragma unroll
  for (int e = 0; e < 16; ++e) {
    int k = k0 + khalf * 16 + e;
    b[e] = (__bf16)p[(size_t)k * ld + col];
  }
  return b;
}

__device__ __forceinline__ float wave_sum32(float v) {
#pragma unroll
  for (int off = 16; off > 0; off >>= 1) v += __shfl_xor(v, off, 32);
  return v;
}

// ---------------- problem constants ----------------
constexpr int NB = 32, NS = 197, ND = 768, NH = 12, NL = 12, NF = 3072, NP = 13;
constexpr int JRP = 232;            // padded K extent of column-major joint (>=224)
constexpr int JCN = 208;            // padded cols (13*16)
constexpr int JE  = JCN * JRP;      // elems per LDS joint buffer (48256)
constexpr int LDY = 776;            // padded row stride for [16][768] bf16 arrays
constexpr int LDH = 3080;           // padded row stride for [16][3072] bf16 array
constexpr int K2_LDS = 49152 + 5 * (16 * LDY * 2) + 12 * 256 * 4 + 16 * LDH * 2; // 284160

// ============================================================================
// Kernel 1: attention rollout (11 chained 197x197 matmuls per (b,h)) + argmax.
// One WG per (b,h). joint kept in LDS as bf16 COLUMN-MAJOR [col][k] (stride
// 232, conflict-free b128), double buffered. Each wave owns a row-tile: its 7
// A fragments are loaded once from HBM (f32->bf16) and reused for 13 column
// tiles; ALL 7 B fragments are preloaded into registers before the WMMA chain
// so the dscnt waits retire incrementally instead of serializing each WMMA.
// Rescale 1/64 per step (argmax-invariant, keeps bf16 magnitudes O(1)).
// ============================================================================
__global__ void __launch_bounds__(256) rollout_kernel(
    const float* __restrict__ att, int* __restrict__ idx_out) {
  extern __shared__ char smem[];
  __bf16* J0 = (__bf16*)smem;
  __bf16* J1 = J0 + JE;
  float*  redv = (float*)(J1 + JE);
  int*    redi = (int*)(redv + 256);

  const int tid   = threadIdx.x;
  const int lane  = tid & 31;
  const int wave  = tid >> 5;
  const int khalf = lane >> 4;
  const int n16   = lane & 15;
  const int bh    = blockIdx.x;                    // b*NH + h
  const size_t layer_stride = (size_t)NB * NH * NS * NS;
  const float* att0 = att + (size_t)bh * NS * NS;  // layer 0 slice

  // zero both buffers (covers padding rows/cols permanently)
  for (int i = tid; i < (2 * JE * 2) / 16; i += 256)
    ((int4*)smem)[i] = make_int4(0, 0, 0, 0);
  __syncthreads();
  // fill J0 column-major: J0[c][r] = att0[r][c] (coalesced global reads)
  for (int i = tid; i < NS * NS; i += 256) {
    int r = i / NS, c = i - r * NS;
    J0[c * JRP + r] = (__bf16)att0[i];
  }
  __syncthreads();

  __bf16* Jc = J0;
  __bf16* Jn = J1;
  for (int n = 1; n < NL; ++n) {
    const float* A = att + (size_t)n * layer_stride + (size_t)bh * (NS * NS);

    // prefetch next layer's slice while we chew on this one (global_prefetch_b8)
    const float* An = A + ((n + 1 < NL) ? layer_stride : 0);
    for (int off = tid * 32; off < NS * NS; off += 256 * 32)
      __builtin_prefetch(An + off, 0, 3);

    for (int ti = wave; ti < 13; ti += 8) {
      const int row = ti * 16 + n16;
      v16bf a[7];
      if (ti < 12) {
#pragma unroll
        for (int kt = 0; kt < 6; ++kt)
          a[kt] = load_a_g32_fast(A, NS, row, kt * 32, khalf);
        a[6] = load_a_g32_guard(A, NS, NS, NS, row, 6 * 32, khalf);
      } else {
#pragma unroll
        for (int kt = 0; kt < 7; ++kt)
          a[kt] = load_a_g32_guard(A, NS, NS, NS, row, kt * 32, khalf);
      }
      for (int tj = 0; tj < 13; ++tj) {
        const __bf16* bcol = Jc + (tj * 16 + n16) * JRP + khalf * 16;
        // preload the whole B set: 14 ds_load_b128 in flight before any WMMA
        v16bf bfr[7];
#pragma unroll
        for (int kt = 0; kt < 7; ++kt) bfr[kt] = load_b_contig(bcol + kt * 32);
        v8f acc = {};
#pragma unroll
        for (int kt = 0; kt < 7; ++kt) acc = wmma_bf16f32(a[kt], bfr[kt], acc);
        union { uint4 q; __bf16 h[8]; } spk;
#pragma unroll
        for (int r = 0; r < 8; ++r) spk.h[r] = (__bf16)(acc[r] * 0.015625f);
        *(uint4*)(Jn + (tj * 16 + n16) * JRP + ti * 16 + khalf * 8) = spk.q;
      }
    }
    __syncthreads();
    __bf16* tmp = Jc; Jc = Jn; Jn = tmp;
  }

  // argmax over CLS row (k==0 of each column), columns 1..196
  float bestv = -1e30f; int besti = 1;
  for (int c = 1 + tid; c < NS; c += 256) {
    float v = (float)Jc[c * JRP];
    if (v > bestv) { bestv = v; besti = c; }
  }
  redv[tid] = bestv; redi[tid] = besti;
  __syncthreads();
  if (tid == 0) {
    float bv = redv[0]; int bi = redi[0];
    for (int i = 1; i < 256; ++i)
      if (redv[i] > bv) { bv = redv[i]; bi = redi[i]; }
    idx_out[bh] = bi;
  }
}

// ============================================================================
// Kernel 2: gather 13 tokens, full ViT block with bf16 WMMA GEMMs.
// One WG (256 threads = 8 waves) per batch. LDS working set ~284KB; bf16
// row-major arrays padded to stride 776/3080 so b128 A-fragment loads are
// bank-conflict-free. K-loops are chunked by 4: 4 B fragments (64 global
// loads) + 4 A fragments are preloaded per group, then 4 WMMAs consume them.
// ============================================================================
__global__ void __launch_bounds__(256) vit_block_kernel(
    const float* __restrict__ vit, const int* __restrict__ idx,
    const float* __restrict__ ln1g, const float* __restrict__ ln1b,
    const float* __restrict__ wq, const float* __restrict__ bq,
    const float* __restrict__ wk, const float* __restrict__ bk,
    const float* __restrict__ wv, const float* __restrict__ bv,
    const float* __restrict__ wo, const float* __restrict__ bo,
    const float* __restrict__ ln2g, const float* __restrict__ ln2b,
    const float* __restrict__ w1, const float* __restrict__ b1,
    const float* __restrict__ w2, const float* __restrict__ b2,
    const float* __restrict__ lnpg, const float* __restrict__ lnpb,
    float* __restrict__ out_states, float* __restrict__ out_probs) {
  extern __shared__ char smem[];
  float*  xf  = (float*)smem;                 // [16][768] f32 residual stream
  __bf16* ybf = (__bf16*)(xf + 16 * ND);      // [16][LDY] LN output (A operand)
  __bf16* qb  = ybf + 16 * LDY;               // [16][LDY]
  __bf16* kb  = qb + 16 * LDY;
  __bf16* vb  = kb + 16 * LDY;
  float*  sc  = (float*)(vb + 16 * LDY);      // [12][16][16] scores/probs f32
  __bf16* cx  = (__bf16*)(sc + NH * 256);     // [16][LDY] attention context
  __bf16* h1  = cx + 16 * LDY;                // [16][LDH] MLP hidden

  const int b     = blockIdx.x;
  const int tid   = threadIdx.x;
  const int lane  = tid & 31;
  const int wave  = tid >> 5;
  const int khalf = lane >> 4;
  const int n16   = lane & 15;

  // zero the whole LDS working set (provides all M/K zero padding)
  for (int i = tid; i < K2_LDS / 16; i += 256) ((int4*)smem)[i] = make_int4(0, 0, 0, 0);
  __syncthreads();

  // gather: row 0 = CLS token, rows 1..12 = per-head top tokens
  for (int i = tid; i < NP * ND; i += 256) {
    int r = i / ND, c = i - r * ND;
    int tok = (r == 0) ? 0 : idx[b * NH + (r - 1)];
    xf[r * ND + c] = vit[((size_t)b * NS + tok) * ND + c];
  }
  __syncthreads();

  // ---- LN1 -> ybf (bf16), wave per row, shuffle reduction ----
  for (int r = wave; r < NP; r += 8) {
    float s = 0.f, s2 = 0.f;
    for (int c = lane; c < ND; c += 32) { float v = xf[r * ND + c]; s += v; s2 += v * v; }
    s = wave_sum32(s); s2 = wave_sum32(s2);
    float mu = s / (float)ND;
    float rs = rsqrtf(s2 / (float)ND - mu * mu + 1e-6f);
    for (int c = lane; c < ND; c += 32)
      ybf[r * LDY + c] = (__bf16)((xf[r * ND + c] - mu) * rs * ln1g[c] + ln1b[c]);
  }
  __syncthreads();

  // ---- Q/K/V projections: 3 x 48 N-tiles, K = 768 (chunked 4-deep) ----
  for (int job = wave; job < 3 * 48; job += 8) {
    int m = job / 48, j = job - m * 48;
    const float* W   = (m == 0) ? wq : (m == 1) ? wk : wv;
    const float* bia = (m == 0) ? bq : (m == 1) ? bk : bv;
    __bf16*      O   = (m == 0) ? qb : (m == 1) ? kb : vb;
    int col = j * 16 + n16;
    v8f acc = {};
    for (int kc = 0; kc < 24; kc += 4) {
      v16bf bfr[4], afr[4];
#pragma unroll
      for (int u = 0; u < 4; ++u) bfr[u] = load_b_g32(W, ND, (kc + u) * 32, col, khalf);
#pragma unroll
      for (int u = 0; u < 4; ++u) afr[u] = load_a_lds16(ybf + n16 * LDY, (kc + u) * 32, khalf);
#pragma unroll
      for (int u = 0; u < 4; ++u) acc = wmma_bf16f32(afr[u], bfr[u], acc);
    }
#pragma unroll
    for (int r = 0; r < 8; ++r) {
      int row = khalf * 8 + r;
      if (row < NP) O[row * LDY + col] = (__bf16)(acc[r] + bia[col]);
    }
  }
  __syncthreads();

  // ---- scores = q @ k^T / 8 per head (K = 64 -> 2 WMMA) ----
  for (int h = wave; h < NH; h += 8) {
    v8f acc = {};
#pragma unroll
    for (int kt = 0; kt < 2; ++kt) {
      v16bf af = load_a_lds16(qb + n16 * LDY, h * 64 + kt * 32, khalf);
      // B[k][n] = k[n][kk]; kk contiguous within a row of kb -> b128 loads
      v16bf bf = load_b_contig(kb + n16 * LDY + h * 64 + kt * 32 + khalf * 16);
      acc = wmma_bf16f32(af, bf, acc);
    }
#pragma unroll
    for (int r = 0; r < 8; ++r)
      sc[h * 256 + (khalf * 8 + r) * 16 + n16] = acc[r] * 0.125f;
  }
  __syncthreads();

  // ---- softmax rows (f32), also emit probs output ----
  for (int t = tid; t < NH * NP; t += 256) {
    int h = t / NP, r = t - h * NP;
    float* row = sc + h * 256 + r * 16;
    float mx = row[0];
#pragma unroll
    for (int c = 1; c < NP; ++c) mx = fmaxf(mx, row[c]);
    float e_[NP]; float s = 0.f;
#pragma unroll
    for (int c = 0; c < NP; ++c) { e_[c] = expf(row[c] - mx); s += e_[c]; }
    float inv = 1.f / s;
#pragma unroll
    for (int c = 0; c < NP; ++c) {
      float p = e_[c] * inv;
      row[c] = p;
      out_probs[(((size_t)b * NH + h) * NP + r) * NP + c] = p;
    }
  }
  __syncthreads();

  // ---- ctx = probs @ v per head (single K-tile of 32, valid K<13) ----
  for (int job = wave; job < NH * 4; job += 8) {
    int h = job / 4, nt = job - h * 4;
    v16bf af;
#pragma unroll
    for (int e = 0; e < 16; ++e) {
      int k = khalf * 16 + e;
      bool ok = (k < NP) && (n16 < NP);
      float v = sc[h * 256 + (n16 < NP ? n16 : 0) * 16 + (k < NP ? k : 0)];
      af[e] = (__bf16)(ok ? v : 0.f);
    }
    int col = h * 64 + nt * 16 + n16;
    v16bf bf;
#pragma unroll
    for (int e = 0; e < 16; ++e) {
      int k = khalf * 16 + e;
      __bf16 v = vb[(k < NP ? k : 0) * LDY + col];
      bf[e] = (k < NP) ? v : (__bf16)0.f;
    }
    v8f acc = {};
    acc = wmma_bf16f32(af, bf, acc);
#pragma unroll
    for (int r = 0; r < 8; ++r) {
      int row = khalf * 8 + r;
      if (row < NP) cx[row * LDY + col] = (__bf16)acc[r];
    }
  }
  __syncthreads();

  // ---- output projection + residual: xf = ctx@wo + bo + xf ----
  for (int j = wave; j < 48; j += 8) {
    int col = j * 16 + n16;
    v8f acc = {};
    for (int kc = 0; kc < 24; kc += 4) {
      v16bf bfr[4], afr[4];
#pragma unroll
      for (int u = 0; u < 4; ++u) bfr[u] = load_b_g32(wo, ND, (kc + u) * 32, col, khalf);
#pragma unroll
      for (int u = 0; u < 4; ++u) afr[u] = load_a_lds16(cx + n16 * LDY, (kc + u) * 32, khalf);
#pragma unroll
      for (int u = 0; u < 4; ++u) acc = wmma_bf16f32(afr[u], bfr[u], acc);
    }
#pragma unroll
    for (int r = 0; r < 8; ++r) {
      int row = khalf * 8 + r;
      if (row < NP) xf[row * ND + col] += acc[r] + bo[col];
    }
  }
  __syncthreads();

  // ---- LN2 -> ybf ----
  for (int r = wave; r < NP; r += 8) {
    float s = 0.f, s2 = 0.f;
    for (int c = lane; c < ND; c += 32) { float v = xf[r * ND + c]; s += v; s2 += v * v; }
    s = wave_sum32(s); s2 = wave_sum32(s2);
    float mu = s / (float)ND;
    float rs = rsqrtf(s2 / (float)ND - mu * mu + 1e-6f);
    for (int c = lane; c < ND; c += 32)
      ybf[r * LDY + c] = (__bf16)((xf[r * ND + c] - mu) * rs * ln2g[c] + ln2b[c]);
  }
  __syncthreads();

  // ---- MLP fc1 + exact GELU -> h1 (bf16) ----
  for (int j = wave; j < NF / 16; j += 8) {
    int col = j * 16 + n16;
    v8f acc = {};
    for (int kc = 0; kc < 24; kc += 4) {
      v16bf bfr[4], afr[4];
#pragma unroll
      for (int u = 0; u < 4; ++u) bfr[u] = load_b_g32(w1, NF, (kc + u) * 32, col, khalf);
#pragma unroll
      for (int u = 0; u < 4; ++u) afr[u] = load_a_lds16(ybf + n16 * LDY, (kc + u) * 32, khalf);
#pragma unroll
      for (int u = 0; u < 4; ++u) acc = wmma_bf16f32(afr[u], bfr[u], acc);
    }
#pragma unroll
    for (int r = 0; r < 8; ++r) {
      int row = khalf * 8 + r;
      if (row < NP) {
        float a = acc[r] + b1[col];
        h1[row * LDH + col] = (__bf16)(0.5f * a * (1.f + erff(a * 0.70710678118654752f)));
      }
    }
  }
  __syncthreads();

  // ---- MLP fc2 + residual: xf = h1@w2 + b2 + xf ----
  for (int j = wave; j < 48; j += 8) {
    int col = j * 16 + n16;
    v8f acc = {};
    for (int kc = 0; kc < NF / 32; kc += 4) {
      v16bf bfr[4], afr[4];
#pragma unroll
      for (int u = 0; u < 4; ++u) bfr[u] = load_b_g32(w2, ND, (kc + u) * 32, col, khalf);
#pragma unroll
      for (int u = 0; u < 4; ++u) afr[u] = load_a_lds16(h1 + n16 * LDH, (kc + u) * 32, khalf);
#pragma unroll
      for (int u = 0; u < 4; ++u) acc = wmma_bf16f32(afr[u], bfr[u], acc);
    }
#pragma unroll
    for (int r = 0; r < 8; ++r) {
      int row = khalf * 8 + r;
      if (row < NP) xf[row * ND + col] += acc[r] + b2[col];
    }
  }
  __syncthreads();

  // ---- final LN -> part_states output ----
  for (int r = wave; r < NP; r += 8) {
    float s = 0.f, s2 = 0.f;
    for (int c = lane; c < ND; c += 32) { float v = xf[r * ND + c]; s += v; s2 += v * v; }
    s = wave_sum32(s); s2 = wave_sum32(s2);
    float mu = s / (float)ND;
    float rs = rsqrtf(s2 / (float)ND - mu * mu + 1e-6f);
    for (int c = lane; c < ND; c += 32)
      out_states[((size_t)b * NP + r) * ND + c] =
          (xf[r * ND + c] - mu) * rs * lnpg[c] + lnpb[c];
  }
}

// ============================================================================
extern "C" void kernel_launch(void* const* d_in, const int* in_sizes, int n_in,
                              void* d_out, int out_size, void* d_ws, size_t ws_size,
                              hipStream_t stream) {
  const float* vit  = (const float*)d_in[0];
  const float* att  = (const float*)d_in[1];
  const float* ln1g = (const float*)d_in[2];
  const float* ln1b = (const float*)d_in[3];
  const float* wq   = (const float*)d_in[4];
  const float* bq   = (const float*)d_in[5];
  const float* wk   = (const float*)d_in[6];
  const float* bk   = (const float*)d_in[7];
  const float* wv   = (const float*)d_in[8];
  const float* bv   = (const float*)d_in[9];
  const float* wo   = (const float*)d_in[10];
  const float* bo   = (const float*)d_in[11];
  const float* ln2g = (const float*)d_in[12];
  const float* ln2b = (const float*)d_in[13];
  const float* w1   = (const float*)d_in[14];
  const float* b1   = (const float*)d_in[15];
  const float* w2   = (const float*)d_in[16];
  const float* b2   = (const float*)d_in[17];
  const float* lnpg = (const float*)d_in[18];
  const float* lnpb = (const float*)d_in[19];

  float* out_states = (float*)d_out;                       // [B,13,768]
  float* out_probs  = out_states + (size_t)NB * NP * ND;   // [B,12,13,13]
  int*   idx_ws     = (int*)d_ws;                          // [B*H]

  const size_t lds1 = (size_t)2 * JE * 2 + 256 * 4 + 256 * 4;  // 195,072 B
  const size_t lds2 = K2_LDS;                                  // 284,160 B
  hipFuncSetAttribute((const void*)rollout_kernel,
                      hipFuncAttributeMaxDynamicSharedMemorySize, (int)lds1);
  hipFuncSetAttribute((const void*)vit_block_kernel,
                      hipFuncAttributeMaxDynamicSharedMemorySize, (int)lds2);

  rollout_kernel<<<NB * NH, 256, lds1, stream>>>(att, idx_ws);
  vit_block_kernel<<<NB, 256, lds2, stream>>>(
      vit, idx_ws, ln1g, ln1b, wq, bq, wk, bk, wv, bv, wo, bo,
      ln2g, ln2b, w1, b1, w2, b2, lnpg, lnpb, out_states, out_probs);

  (void)in_sizes; (void)n_in; (void)out_size; (void)ws_size;
}
